// seq2seq_10746008174981
// MI455X (gfx1250) — matrix operand order repaired and checked by reading, not compiled, not run on hardware
//
#include <hip/hip_runtime.h>
#include <hip/hip_bf16.h>

// ---------------- dims ----------------
#define Vv   30000
#define Ee   256
#define Hh   512
#define H3   1536
#define Bb   64
#define Ss   64
#define Tt   32
#define KFC  1024   // 2H
#define KRNN 768    // E+H

typedef __attribute__((ext_vector_type(8)))  float          v8f;
typedef __attribute__((ext_vector_type(2)))  float          v2f;
typedef __attribute__((ext_vector_type(16))) __bf16         v16bf;
typedef __attribute__((ext_vector_type(8)))  unsigned short us8;

// =====================================================================
// Large fp32 GEMM:  C[M x N] = A[M x K] @ W[N x K]^T + bias
// V_WMMA_F32_16X16X4_F32. Block = 256 thr (8 waves), tile 64x64.
// Used for the big batched GEMMs (M = S*B = 4096).
// =====================================================================
__global__ void k_gemm_f32(const float* __restrict__ A, int lda,
                           const float* __restrict__ W,            // N x K row-major
                           const float* __restrict__ bias,
                           float* __restrict__ C, int ldc, int K)
{
    const int lane  = threadIdx.x & 31;
    const int wave  = threadIdx.x >> 5;
    const int l16   = lane & 15;
    const int khalf = lane >> 4;                 // 0 or 1 (upper-lane K half)
    const int m0    = blockIdx.y * 64 + (wave & 3) * 16;
    const int nb    = blockIdx.x * 64 + (wave >> 2) * 32;

    const float* Arow  = A + (size_t)(m0 + l16) * lda + 2 * khalf;
    const float* Wrow0 = W + (size_t)(nb + l16) * K   + 2 * khalf;
    const float* Wrow1 = Wrow0 + (size_t)16 * K;

    v8f acc0 = {}; v8f acc1 = {};
    #pragma unroll 4
    for (int k = 0; k < K; k += 4) {
        v2f a  = *(const v2f*)(Arow  + k);       // A frag: 16x4 f32
        v2f b0 = *(const v2f*)(Wrow0 + k);       // B frag: 4x16 f32 (W^T)
        v2f b1 = *(const v2f*)(Wrow1 + k);
        acc0 = __builtin_amdgcn_wmma_f32_16x16x4_f32(false, a, false, b0,
                                                     (short)0, acc0, false, false);
        acc1 = __builtin_amdgcn_wmma_f32_16x16x4_f32(false, a, false, b1,
                                                     (short)0, acc1, false, false);
    }
    const int col0 = nb + l16, col1 = col0 + 16;
    const float bb0 = bias ? bias[col0] : 0.f;
    const float bb1 = bias ? bias[col1] : 0.f;
    #pragma unroll
    for (int i = 0; i < 8; ++i) {                // C layout: VGPR i -> M=i (+8 upper lanes)
        const int m = m0 + i + 8 * khalf;
        C[(size_t)m * ldc + col0] = acc0[i] + bb0;
        C[(size_t)m * ldc + col1] = acc1[i] + bb1;
    }
}

// =====================================================================
// Serial-chain fp32 GEMM, M fixed to 64 (the whole batch):
//   one wave per block; wave owns all 4 m-tiles x 16 columns.
//   -> each W fragment loaded exactly once, grid = N/16 blocks for
//      4x more parallelism on the latency-critical recurrence GEMMs.
// =====================================================================
__global__ void k_gemm_f32_w16(const float* __restrict__ A, int lda,   // 64 x K
                               const float* __restrict__ W,            // N x K
                               const float* __restrict__ bias,
                               float* __restrict__ C, int ldc, int K)
{
    const int lane  = threadIdx.x & 31;
    const int l16   = lane & 15;
    const int khalf = lane >> 4;
    const int n0    = blockIdx.x * 16;

    const float* Wrow = W + (size_t)(n0 + l16) * K + 2 * khalf;
    const float* Ar0  = A + (size_t)(0  + l16) * lda + 2 * khalf;
    const float* Ar1  = A + (size_t)(16 + l16) * lda + 2 * khalf;
    const float* Ar2  = A + (size_t)(32 + l16) * lda + 2 * khalf;
    const float* Ar3  = A + (size_t)(48 + l16) * lda + 2 * khalf;

    v8f a0 = {}, a1 = {}, a2 = {}, a3 = {};
    #pragma unroll 4
    for (int k = 0; k < K; k += 4) {
        v2f b  = *(const v2f*)(Wrow + k);        // shared across all 4 m-tiles
        v2f f0 = *(const v2f*)(Ar0 + k);
        v2f f1 = *(const v2f*)(Ar1 + k);
        v2f f2 = *(const v2f*)(Ar2 + k);
        v2f f3 = *(const v2f*)(Ar3 + k);
        a0 = __builtin_amdgcn_wmma_f32_16x16x4_f32(false, f0, false, b, (short)0, a0, false, false);
        a1 = __builtin_amdgcn_wmma_f32_16x16x4_f32(false, f1, false, b, (short)0, a1, false, false);
        a2 = __builtin_amdgcn_wmma_f32_16x16x4_f32(false, f2, false, b, (short)0, a2, false, false);
        a3 = __builtin_amdgcn_wmma_f32_16x16x4_f32(false, f3, false, b, (short)0, a3, false, false);
    }
    const int col = n0 + l16;
    const float bb = bias ? bias[col] : 0.f;
    #pragma unroll
    for (int i = 0; i < 8; ++i) {
        const int r = i + 8 * khalf;
        C[(size_t)(r     ) * ldc + col] = a0[i] + bb;
        C[(size_t)(r + 16) * ldc + col] = a1[i] + bb;
        C[(size_t)(r + 32) * ldc + col] = a2[i] + bb;
        C[(size_t)(r + 48) * ldc + col] = a3[i] + bb;
    }
}

// =====================================================================
// fc GEMM (bf16 weights, fp32 accumulate), one wave per block:
//   wave owns all 64 batch rows x 16 vocab columns, so the bandwidth-
//   critical weight fragment is fetched exactly once (r = 1 traffic).
//   out[(b*T+t)*V + v] = hc[b,:] . fcW[v,:] + fc_b[v]
// =====================================================================
__global__ void k_fc_bf16(const unsigned short* __restrict__ hc,   // 64 x 1024 bf16
                          const unsigned short* __restrict__ Wbf,  // V x 1024 bf16
                          const float* __restrict__ bias,
                          float* __restrict__ out, int t)
{
    const int lane = threadIdx.x & 31;
    const int l16  = lane & 15;
    const int half = lane >> 4;
    const int n0   = blockIdx.x * 16;

    const unsigned short* brow = Wbf + (size_t)(n0 + l16) * KFC + 8 * half;
    const unsigned short* ar0  = hc + (size_t)(0  + l16) * KFC + 8 * half;
    const unsigned short* ar1  = hc + (size_t)(16 + l16) * KFC + 8 * half;
    const unsigned short* ar2  = hc + (size_t)(32 + l16) * KFC + 8 * half;
    const unsigned short* ar3  = hc + (size_t)(48 + l16) * KFC + 8 * half;

    v8f c0 = {}, c1 = {}, c2 = {}, c3 = {};
    #pragma unroll 2
    for (int k = 0; k < KFC; k += 32) {
        __builtin_prefetch(brow + k + 256, 0, 1);        // global_prefetch_b8
        union { v16bf v; us8 h[2]; } b, a0, a1, a2, a3;
        b.h[0]  = *(const us8*)(brow + k);               // B 32x16: k 8h..8h+7, 16+8h..16+8h+7
        b.h[1]  = *(const us8*)(brow + k + 16);
        a0.h[0] = *(const us8*)(ar0 + k);  a0.h[1] = *(const us8*)(ar0 + k + 16);
        a1.h[0] = *(const us8*)(ar1 + k);  a1.h[1] = *(const us8*)(ar1 + k + 16);
        a2.h[0] = *(const us8*)(ar2 + k);  a2.h[1] = *(const us8*)(ar2 + k + 16);
        a3.h[0] = *(const us8*)(ar3 + k);  a3.h[1] = *(const us8*)(ar3 + k + 16);
        c0 = __builtin_amdgcn_wmma_f32_16x16x32_bf16(false, a0.v, false, b.v, (short)0, c0, false, false);
        c1 = __builtin_amdgcn_wmma_f32_16x16x32_bf16(false, a1.v, false, b.v, (short)0, c1, false, false);
        c2 = __builtin_amdgcn_wmma_f32_16x16x32_bf16(false, a2.v, false, b.v, (short)0, c2, false, false);
        c3 = __builtin_amdgcn_wmma_f32_16x16x32_bf16(false, a3.v, false, b.v, (short)0, c3, false, false);
    }
    const int v  = n0 + l16;
    const float bb = bias[v];
    #pragma unroll
    for (int i = 0; i < 8; ++i) {
        const int r = i + 8 * half;
        out[((size_t)(r     ) * Tt + t) * Vv + v] = c0[i] + bb;
        out[((size_t)(r + 16) * Tt + t) * Vv + v] = c1[i] + bb;
        out[((size_t)(r + 32) * Tt + t) * Vv + v] = c2[i] + bb;
        out[((size_t)(r + 48) * Tt + t) * Vv + v] = c3[i] + bb;
    }
}

// ---------------- elementwise / gather helpers ----------------
__global__ void k_zero(float* __restrict__ p, int n) {
    int g = blockIdx.x * blockDim.x + threadIdx.x;
    if (g < n) p[g] = 0.f;
}
__global__ void k_copy(const float* __restrict__ s, float* __restrict__ d, int n) {
    int g = blockIdx.x * blockDim.x + threadIdx.x;
    if (g < n) d[g] = s[g];
}
__global__ void k_cvt_bf16(const float* __restrict__ s, __hip_bfloat16* __restrict__ d, int n) {
    int g = blockIdx.x * blockDim.x + threadIdx.x;
    if (g < n) d[g] = __float2bfloat16(s[g]);
}
// ex[(t*B+b)*E + e] = emb[x[b,t]*E + e]
__global__ void k_embed_x(const int* __restrict__ x, const float* __restrict__ emb,
                          float* __restrict__ ex) {
    int g = blockIdx.x * blockDim.x + threadIdx.x;       // over S*B*E
    int e = g % Ee, r = g / Ee;
    int b = r % Bb, t = r / Bb;
    ex[g] = emb[(size_t)x[b * Ss + t] * Ee + e];
}
__global__ void k_embed_start(const int* __restrict__ st, const float* __restrict__ emb,
                              float* __restrict__ de) {
    int g = blockIdx.x * blockDim.x + threadIdx.x;       // over B*E
    int e = g % Ee, b = g / Ee;
    de[g] = emb[(size_t)st[b] * Ee + e];
}
// GRU gate math; gi/gh are B x 3H (biases already folded in by GEMM)
__global__ void k_gates(const float* __restrict__ gi, const float* __restrict__ gh,
                        float* __restrict__ h, float* __restrict__ out /*nullable*/) {
    int g = blockIdx.x * blockDim.x + threadIdx.x;       // over B*H
    int j = g % Hh, b = g / Hh;
    const float* gib = gi + (size_t)b * H3;
    const float* ghb = gh + (size_t)b * H3;
    float r  = 1.f / (1.f + expf(-(gib[j] + ghb[j])));
    float z  = 1.f / (1.f + expf(-(gib[Hh + j] + ghb[Hh + j])));
    float n  = tanhf(gib[2 * Hh + j] + r * ghb[2 * Hh + j]);
    float hn = (1.f - z) * n + z * h[g];
    h[g] = hn;
    if (out) out[g] = hn;
}
__global__ void k_concat_rnn(const float* __restrict__ de, const float* __restrict__ ctx,
                             float* __restrict__ ri) {
    int g = blockIdx.x * blockDim.x + threadIdx.x;       // over B*768
    int j = g % KRNN, b = g / KRNN;
    ri[g] = (j < Ee) ? de[b * Ee + j] : ctx[b * Hh + (j - Ee)];
}
__global__ void k_concat_hc(const float* __restrict__ h, const float* __restrict__ ctx,
                            __hip_bfloat16* __restrict__ hc) {
    int g = blockIdx.x * blockDim.x + threadIdx.x;       // over B*1024
    int j = g % KFC, b = g / KFC;
    float v = (j < Hh) ? h[b * Hh + j] : ctx[b * Hh + (j - Hh)];
    hc[g] = __float2bfloat16(v);
}
// block per batch: argmax over V of scores row, then gather embedding row
__global__ void k_argmax_embed(const float* __restrict__ out, int t,
                               const float* __restrict__ emb, float* __restrict__ de) {
    __shared__ float sv[256]; __shared__ int si[256];
    const int b = blockIdx.x, tid = threadIdx.x;
    const float* row = out + ((size_t)b * Tt + t) * Vv;
    float best = -3.4e38f; int bi = 0;
    for (int v = tid; v < Vv; v += 256) {
        float s = row[v];
        if (s > best) { best = s; bi = v; }                 // first occurrence wins
    }
    sv[tid] = best; si[tid] = bi; __syncthreads();
    for (int off = 128; off > 0; off >>= 1) {
        if (tid < off) {
            float o = sv[tid + off]; int oi = si[tid + off];
            if (o > sv[tid] || (o == sv[tid] && oi < si[tid])) { sv[tid] = o; si[tid] = oi; }
        }
        __syncthreads();
    }
    const int dec = si[0];
    de[b * Ee + tid] = emb[(size_t)dec * Ee + tid];         // E == 256 == blockDim
}
// block per batch: alpha = softmax_s(temp[s,b,:].h[b,:]); ctx = sum_s alpha*enc_outs[s,b,:]
__global__ void k_attn(const float* __restrict__ temp, const float* __restrict__ eo,
                       const float* __restrict__ h, float* __restrict__ ctx) {
    __shared__ float red[256]; __shared__ float logit[Ss]; __shared__ float inv;
    const int b = blockIdx.x, tid = threadIdx.x;
    const int s = tid >> 2, q = tid & 3;
    const float* tp = temp + ((size_t)s * Bb + b) * Hh;
    const float* hb = h + (size_t)b * Hh;
    float p = 0.f;
    for (int j = q * 128; j < q * 128 + 128; ++j) p += tp[j] * hb[j];
    red[tid] = p; __syncthreads();
    if (q == 0) logit[s] = red[tid] + red[tid + 1] + red[tid + 2] + red[tid + 3];
    __syncthreads();
    if (tid == 0) {
        float m = logit[0];
        for (int i = 1; i < Ss; ++i) m = fmaxf(m, logit[i]);
        float ss = 0.f;
        for (int i = 0; i < Ss; ++i) { float e = expf(logit[i] - m); logit[i] = e; ss += e; }
        inv = 1.f / ss;
    }
    __syncthreads();
    for (int j = tid; j < Hh; j += 256) {
        float c = 0.f;
        #pragma unroll 8
        for (int s2 = 0; s2 < Ss; ++s2)
            c += logit[s2] * eo[((size_t)s2 * Bb + b) * Hh + j];
        ctx[(size_t)b * Hh + j] = c * inv;
    }
}

// =====================================================================
static inline size_t al256(size_t x) { return (x + 255) & ~(size_t)255; }

extern "C" void kernel_launch(void* const* d_in, const int* in_sizes, int n_in,
                              void* d_out, int out_size, void* d_ws, size_t ws_size,
                              hipStream_t stream) {
    (void)in_sizes; (void)n_in; (void)out_size; (void)ws_size;
    const int*   x        = (const int*)  d_in[0];
    const int*   start    = (const int*)  d_in[1];
    const float* emb      = (const float*)d_in[3];
    const float* enc_Wih  = (const float*)d_in[4];
    const float* enc_Whh  = (const float*)d_in[5];
    const float* enc_bih  = (const float*)d_in[6];
    const float* enc_bhh  = (const float*)d_in[7];
    const float* dec_Wih  = (const float*)d_in[8];
    const float* dec_Whh  = (const float*)d_in[9];
    const float* dec_bih  = (const float*)d_in[10];
    const float* dec_bhh  = (const float*)d_in[11];
    const float* fc_W     = (const float*)d_in[12];
    const float* fc_b     = (const float*)d_in[13];
    const float* attn_W   = (const float*)d_in[14];
    const float* attn_b   = (const float*)d_in[15];
    float* out = (float*)d_out;

    // ---- workspace carve-up ----
    char* ws = (char*)d_ws; size_t cur = 0;
    auto take = [&](size_t bytes) { char* p = ws + cur; cur += al256(bytes); return p; };
    float* ex      = (float*)take((size_t)Ss * Bb * Ee * 4);      // (S*B, E)
    float* gi_all  = (float*)take((size_t)Ss * Bb * H3 * 4);      // (S*B, 3H)
    float* eo      = (float*)take((size_t)Ss * Bb * Hh * 4);      // enc_outs (S,B,H)
    float* temp    = (float*)take((size_t)Ss * Bb * Hh * 4);
    __hip_bfloat16* fcWbf = (__hip_bfloat16*)take((size_t)Vv * KFC * 2);
    float* h       = (float*)take((size_t)Bb * Hh * 4);
    float* ctx     = (float*)take((size_t)Bb * Hh * 4);
    float* de      = (float*)take((size_t)Bb * Ee * 4);
    float* ri      = (float*)take((size_t)Bb * KRNN * 4);
    float* GI      = (float*)take((size_t)Bb * H3 * 4);
    float* GH      = (float*)take((size_t)Bb * H3 * 4);
    __hip_bfloat16* hc = (__hip_bfloat16*)take((size_t)Bb * KFC * 2);

    // ---- one-time prep ----
    k_cvt_bf16<<<(Vv * KFC) / 256, 256, 0, stream>>>(fc_W, fcWbf, Vv * KFC);
    k_embed_x<<<(Ss * Bb * Ee) / 256, 256, 0, stream>>>(x, emb, ex);
    // GI_all = ex @ enc_Wih^T + bih   (M=4096, N=1536, K=256)
    k_gemm_f32<<<dim3(H3 / 64, (Ss * Bb) / 64), 256, 0, stream>>>(
        ex, Ee, enc_Wih, enc_bih, gi_all, H3, Ee);
    k_zero<<<(Bb * Hh) / 256, 256, 0, stream>>>(h, Bb * Hh);

    // ---- encoder (serial in t) ----
    for (int t = 0; t < Ss; ++t) {
        k_gemm_f32_w16<<<H3 / 16, 32, 0, stream>>>(
            h, Hh, enc_Whh, enc_bhh, GH, H3, Hh);
        k_gates<<<(Bb * Hh) / 256, 256, 0, stream>>>(
            gi_all + (size_t)t * Bb * H3, GH, h, eo + (size_t)t * Bb * Hh);
    }
    // temp = enc_outs @ attn_W^T + attn_b   (M=4096, N=512, K=512)
    k_gemm_f32<<<dim3(Hh / 64, (Ss * Bb) / 64), 256, 0, stream>>>(
        eo, Hh, attn_W, attn_b, temp, Hh, Hh);
    k_copy<<<(Bb * Hh) / 256, 256, 0, stream>>>(h, ctx, Bb * Hh);        // context0 = h_enc
    k_embed_start<<<(Bb * Ee) / 256, 256, 0, stream>>>(start, emb, de);

    // ---- decoder (serial in t) ----
    for (int t = 0; t < Tt; ++t) {
        k_concat_rnn<<<(Bb * KRNN) / 256, 256, 0, stream>>>(de, ctx, ri);
        k_gemm_f32_w16<<<H3 / 16, 32, 0, stream>>>(
            ri, KRNN, dec_Wih, dec_bih, GI, H3, KRNN);
        k_gemm_f32_w16<<<H3 / 16, 32, 0, stream>>>(
            h, Hh, dec_Whh, dec_bhh, GH, H3, Hh);
        k_gates<<<(Bb * Hh) / 256, 256, 0, stream>>>(GI, GH, h, nullptr);
        k_concat_hc<<<(Bb * KFC) / 256, 256, 0, stream>>>(h, ctx, hc);
        k_fc_bf16<<<Vv / 16, 32, 0, stream>>>(
            (const unsigned short*)hc, (const unsigned short*)fcWbf, fc_b, out, t);
        k_argmax_embed<<<Bb, 256, 0, stream>>>(out, t, emb, de);
        k_attn<<<Bb, 256, 0, stream>>>(temp, eo, h, ctx);
    }
}
